// Shuffle_MSA_15917148799465
// MI455X (gfx1250) — compile-verified, matrix operand order checked by reading, not run on hardware
//
#include <hip/hip_runtime.h>
#include <hip/hip_bf16.h>

typedef __attribute__((ext_vector_type(16))) __bf16 v16bf;
typedef __attribute__((ext_vector_type(8)))  __bf16 v8bf;
typedef __attribute__((ext_vector_type(4)))  __bf16 v4bf;
typedef __attribute__((ext_vector_type(2)))  __bf16 v2bf;
typedef __attribute__((ext_vector_type(8)))  float  v8f;
typedef __attribute__((ext_vector_type(4)))  float  v4f;
typedef __attribute__((ext_vector_type(2)))  float  v2f;

#define WMMA_BF16(a, b, c) \
  __builtin_amdgcn_wmma_f32_16x16x32_bf16(false, (a), false, (b), (short)0, (c), false, false)

__device__ inline __bf16 cvt(float f) { return (__bf16)f; }  // native fptrunc on gfx1250

// bf16 fragment from LDS, rows stride `ld` halfs (CDNA5 ISA layout):
//   lanes 0-15 : row = lane,    K in {0..7} U {16..23}
//   lanes 16-31: row = lane-16, K in {8..15} U {24..31}
__device__ inline v16bf load_frag(const __bf16* base, int ld) {
  int lane = threadIdx.x & 31;
  const __bf16* p = base + (lane & 15) * ld + ((lane >> 4) << 3);
  v8bf lo = *(const v8bf*)p;
  v8bf hi = *(const v8bf*)(p + 16);
  return __builtin_shufflevector(lo, hi, 0,1,2,3,4,5,6,7,8,9,10,11,12,13,14,15);
}

// ---------------------------------------------------------------- K0: weights -> bf16
__global__ __launch_bounds__(256) void cvt_weights(const float* __restrict__ wqkv,
                                                   const float* __restrict__ wproj,
                                                   __bf16* __restrict__ wq_bf,
                                                   __bf16* __restrict__ wp_bf) {
  int i4 = (blockIdx.x * 256 + threadIdx.x) * 4;
  if (i4 < 1536 * 512) {
    v4f v = *(const v4f*)(wqkv + i4);
    v4bf p = {cvt(v.x), cvt(v.y), cvt(v.z), cvt(v.w)};
    *(v4bf*)(wq_bf + i4) = p;
  }
  if (i4 < 512 * 512) {
    v4f v = *(const v4f*)(wproj + i4);
    v4bf p = {cvt(v.x), cvt(v.y), cvt(v.z), cvt(v.w)};
    *(v4bf*)(wp_bf + i4) = p;
  }
}

// ---------------------------------------------------------------- K1: QKV GEMM
// Block = 256 thr (8 waves). Tile: M=128 (out channels) x N=128 (one padded image row).
// Wave w: M-strip (w&3)*32, N-strip (w>>2)*64 -> 2x4 WMMA tiles, K in chunks of 32.
__global__ __launch_bounds__(256) void qkv_gemm(const float* __restrict__ x,
                                                const __bf16* __restrict__ wq,
                                                __bf16* __restrict__ qkv) {
  __shared__ __align__(16) __bf16 As[128 * 32];  // [m][k]
  __shared__ __align__(16) __bf16 Bs[128 * 32];  // [pix][k]
  const int h    = blockIdx.x;        // padded row 0..127
  const int m0   = blockIdx.y * 128;  // output-channel base
  const int b    = blockIdx.z;
  const int t    = threadIdx.x;
  const int lane = t & 31, wave = t >> 5;
  const int m_off = (wave & 3) * 32, n_off = (wave >> 2) * 64;

  v8f acc[2][4];
  #pragma unroll
  for (int i = 0; i < 2; ++i)
    #pragma unroll
    for (int j = 0; j < 4; ++j) acc[i][j] = {};

  for (int kk = 0; kk < 512; kk += 32) {
    #pragma unroll
    for (int i = 0; i < 2; ++i) {                     // A: 128x32 bf16, 16B vector copies
      int idx = i * 256 + t;
      int ml = idx >> 2, seg = (idx & 3) * 8;
      *(v8bf*)(As + ml * 32 + seg) =
          *(const v8bf*)(wq + (size_t)(m0 + ml) * 512 + kk + seg);
    }
    #pragma unroll
    for (int i = 0; i < 2; ++i) {                     // B: 4ch x 2pix blocks -> b64 LDS stores
      int idx = i * 256 + t;                          // 0..511
      int cl  = (idx >> 6) * 4;                       // channel group of 4
      int wl2 = (idx & 63) * 2;                       // pixel pair
      v2f v0 = {0.f, 0.f}, v1 = {0.f, 0.f}, v2 = {0.f, 0.f}, v3 = {0.f, 0.f};
      if (h < 126 && wl2 < 126) {
        const float* rb = x + (((size_t)b * 512 + kk + cl) * 126 + h) * 126 + wl2;
        v0 = *(const v2f*)(rb);
        v1 = *(const v2f*)(rb + 126 * 126);
        v2 = *(const v2f*)(rb + 2 * 126 * 126);
        v3 = *(const v2f*)(rb + 3 * 126 * 126);
      }
      v4bf p0 = {cvt(v0.x), cvt(v1.x), cvt(v2.x), cvt(v3.x)};
      v4bf p1 = {cvt(v0.y), cvt(v1.y), cvt(v2.y), cvt(v3.y)};
      *(v4bf*)(Bs + wl2 * 32 + cl)       = p0;
      *(v4bf*)(Bs + (wl2 + 1) * 32 + cl) = p1;
    }
    if (kk + 32 < 512)                                // prefetch next A tile
      __builtin_prefetch(wq + (size_t)(m0 + (t >> 1)) * 512 + kk + 32 + (t & 1) * 16, 0, 1);
    __syncthreads();
    v16bf af[2], bfr[4];
    #pragma unroll
    for (int mt = 0; mt < 2; ++mt) af[mt] = load_frag(&As[(m_off + mt * 16) * 32], 32);
    #pragma unroll
    for (int nt = 0; nt < 4; ++nt) bfr[nt] = load_frag(&Bs[(n_off + nt * 16) * 32], 32);
    #pragma unroll
    for (int mt = 0; mt < 2; ++mt)
      #pragma unroll
      for (int nt = 0; nt < 4; ++nt)
        acc[mt][nt] = WMMA_BF16(af[mt], bfr[nt], acc[mt][nt]);
    __syncthreads();
  }

  const float SCALE = 0.17677669529663688f;  // 32^-0.5, folded into q
  #pragma unroll
  for (int mt = 0; mt < 2; ++mt) {
    int o8 = m0 + m_off + mt * 16 + ((lane >> 4) << 3);  // aligned 8-block of channels
    int qi = o8 >> 9, oc = o8 & 511;
    int head = oc >> 5, d0 = oc & 31;                    // 8 consecutive d's
    float sc = (qi == 0) ? SCALE : 1.0f;
    #pragma unroll
    for (int nt = 0; nt < 4; ++nt) {
      int n   = n_off + nt * 16 + (lane & 15);
      int win = b * 256 + (h >> 3) * 16 + (n >> 3);
      int tok = ((h & 7) << 3) | (n & 7);
      v8bf pk;
      #pragma unroll
      for (int r = 0; r < 8; ++r) pk[r] = cvt(acc[mt][nt][r] * sc);
      *(v8bf*)(qkv + ((((size_t)qi * 2048 + win) * 16 + head) * 64 + tok) * 32 + d0) = pk;
    }
  }
}

// ---------------------------------------------------------------- K2: window attention
// Block = 128 thr (4 waves) per (window, head). Wave w owns score rows [16w,16w+16).
__global__ __launch_bounds__(128) void attn_kernel(const __bf16* __restrict__ qkv,
                                                   const float* __restrict__ btab,
                                                   __bf16* __restrict__ aout) {
  __shared__ __align__(16) __bf16 qs[64 * 32];      // [tok][d]
  __shared__ __align__(16) __bf16 ks[64 * 32];      // [tok][d]  == B layout [N][K]
  __shared__ __align__(16) __bf16 vT[32 * 64];      // [d][tok]  == B layout [N][K]
  __shared__ __align__(16) __bf16 ps[4 * 16 * 64];  // P strips; reused for output staging
  __shared__ float bsh[225];                        // this head's bias column
  const int win = blockIdx.x, head = blockIdx.y;
  const int t = threadIdx.x, lane = t & 31, wave = t >> 5;
  const size_t plane = (size_t)2048 * 16 * 64 * 32;
  const size_t base  = ((size_t)win * 16 + head) * 64 * 32;

  #pragma unroll
  for (int i = 0; i < 2; ++i) {                     // q,k: straight 16B copies
    int e = (i * 128 + t) * 8;
    *(v8bf*)(qs + e) = *(const v8bf*)(qkv + base + e);
    *(v8bf*)(ks + e) = *(const v8bf*)(qkv + base + plane + e);
  }
  {                                                 // v: token-pair pack -> b32 LDS stores
    int tp = t >> 2, seg = t & 3;                   // token pair, d-segment
    int d0 = seg * 8;
    const __bf16* vbase = qkv + base + 2 * plane;
    v8bf lv0 = *(const v8bf*)(vbase + (2 * tp) * 32 + d0);
    v8bf lv1 = *(const v8bf*)(vbase + (2 * tp + 1) * 32 + d0);
    #pragma unroll
    for (int j = 0; j < 8; ++j) {
      v2bf pr = {lv0[j], lv1[j]};
      *(v2bf*)(vT + (d0 + j) * 64 + 2 * tp) = pr;
    }
  }
  for (int e = t; e < 225; e += 128) bsh[e] = btab[e * 16 + head];
  __syncthreads();

  v16bf aq = load_frag(qs + wave * 16 * 32, 32);
  v8f s[4];
  #pragma unroll
  for (int nt = 0; nt < 4; ++nt) {
    v8f z = {};
    s[nt] = WMMA_BF16(aq, load_frag(ks + nt * 16 * 32, 32), z);
  }

  const int nl = lane & 15;
  const int mh = (lane >> 4) << 3;
  #pragma unroll
  for (int nt = 0; nt < 4; ++nt) {                  // relative-position bias from LDS
    int j = nt * 16 + nl;
    int yj = j >> 3, xj = j & 7;
    #pragma unroll
    for (int r = 0; r < 8; ++r) {
      int i = wave * 16 + mh + r;
      int idx = ((i >> 3) - yj + 7) * 15 + ((i & 7) - xj + 7);
      s[nt][r] += bsh[idx];
    }
  }
  float rinv[8];
  #pragma unroll
  for (int r = 0; r < 8; ++r) {                     // softmax: row = 16 lanes of half-wave
    float m = fmaxf(fmaxf(s[0][r], s[1][r]), fmaxf(s[2][r], s[3][r]));
    m = fmaxf(m, __shfl_xor(m, 1));
    m = fmaxf(m, __shfl_xor(m, 2));
    m = fmaxf(m, __shfl_xor(m, 4));
    m = fmaxf(m, __shfl_xor(m, 8));
    float sum = 0.f;
    #pragma unroll
    for (int nt = 0; nt < 4; ++nt) { float e = __expf(s[nt][r] - m); s[nt][r] = e; sum += e; }
    sum += __shfl_xor(sum, 1);
    sum += __shfl_xor(sum, 2);
    sum += __shfl_xor(sum, 4);
    sum += __shfl_xor(sum, 8);
    rinv[r] = 1.0f / sum;
  }
  __bf16* psw = ps + wave * 16 * 64;                // restage P as [m][j] for A-fragments
  #pragma unroll
  for (int nt = 0; nt < 4; ++nt)
    #pragma unroll
    for (int r = 0; r < 8; ++r)
      psw[(mh + r) * 64 + nt * 16 + nl] = cvt(s[nt][r]);
  asm volatile("s_wait_dscnt 0" ::: "memory");      // same-wave LDS RAW across lanes

  v8f o0 = {}, o1 = {};
  #pragma unroll
  for (int kt = 0; kt < 2; ++kt) {
    v16bf ap = load_frag(psw + kt * 32, 64);
    o0 = WMMA_BF16(ap, load_frag(vT + kt * 32, 64), o0);
    o1 = WMMA_BF16(ap, load_frag(vT + 16 * 64 + kt * 32, 64), o1);
  }

  __syncthreads();                                  // all P reads done -> reuse ps
  #pragma unroll
  for (int r = 0; r < 8; ++r) {                     // stage output as [tok][d] (64x32)
    int i = wave * 16 + mh + r;
    ps[i * 32 + nl]      = cvt(o0[r] * rinv[r]);
    ps[i * 32 + 16 + nl] = cvt(o1[r] * rinv[r]);
  }
  __syncthreads();
  const int bb = win >> 8, wi = win & 255;
  const int wh = wi >> 4, wwd = wi & 15;
  #pragma unroll
  for (int i = 0; i < 2; ++i) {                     // coalesced 16B NHWC stores
    int e = (i * 128 + t) * 8;
    int tok = e >> 5, dseg = e & 31;
    int hh = wh * 8 + (tok >> 3), wpix = wwd * 8 + (tok & 7);
    size_t po = ((size_t)bb * 16384 + hh * 128 + wpix) * 512 + head * 32 + dseg;
    *(v8bf*)(aout + po) = *(const v8bf*)(ps + e);
  }
}

// ---------------------------------------------------------------- K3: proj GEMM + bias
__global__ __launch_bounds__(256) void proj_gemm(const __bf16* __restrict__ xin,   // [b][pix][c] bf16
                                                 const __bf16* __restrict__ wp,    // [512][512] bf16
                                                 const float* __restrict__ bp,
                                                 float* __restrict__ out) {
  __shared__ __align__(16) unsigned char smem[16384];
  __bf16* As = (__bf16*)smem;            // [128][32]
  __bf16* Bs = (__bf16*)(smem + 8192);   // [128][32]
  float* stage = (float*)smem;           // reused: [32][128] f32 output transpose
  const int h  = blockIdx.x;
  const int m0 = blockIdx.y * 128;
  const int b  = blockIdx.z;
  const int t  = threadIdx.x;
  const int lane = t & 31, wave = t >> 5;
  const int m_off = (wave & 3) * 32, n_off = (wave >> 2) * 64;

  v8f acc[2][4];
  #pragma unroll
  for (int i = 0; i < 2; ++i)
    #pragma unroll
    for (int j = 0; j < 4; ++j) acc[i][j] = {};

  for (int kk = 0; kk < 512; kk += 32) {
    #pragma unroll
    for (int i = 0; i < 2; ++i) {                    // A: 16B vector copies
      int idx = i * 256 + t;
      int ml = idx >> 2, seg = (idx & 3) * 8;
      *(v8bf*)(As + ml * 32 + seg) =
          *(const v8bf*)(wp + (size_t)(m0 + ml) * 512 + kk + seg);
    }
    #pragma unroll
    for (int i = 0; i < 2; ++i) {                    // B: NHWC contiguous, 16B copies
      int idx = i * 256 + t;
      int wl = idx >> 2, seg = (idx & 3) * 8;
      *(v8bf*)(Bs + wl * 32 + seg) =
          *(const v8bf*)(xin + ((size_t)b * 16384 + h * 128 + wl) * 512 + kk + seg);
    }
    if (kk + 32 < 512)
      __builtin_prefetch(wp + (size_t)(m0 + (t >> 1)) * 512 + kk + 32 + (t & 1) * 16, 0, 1);
    __syncthreads();
    v16bf af[2], bfr[4];
    #pragma unroll
    for (int mt = 0; mt < 2; ++mt) af[mt] = load_frag(&As[(m_off + mt * 16) * 32], 32);
    #pragma unroll
    for (int nt = 0; nt < 4; ++nt) bfr[nt] = load_frag(&Bs[(n_off + nt * 16) * 32], 32);
    #pragma unroll
    for (int mt = 0; mt < 2; ++mt)
      #pragma unroll
      for (int nt = 0; nt < 4; ++nt)
        acc[mt][nt] = WMMA_BF16(af[mt], bfr[nt], acc[mt][nt]);
    __syncthreads();
  }

  // Epilogue: transpose through LDS in 4 chunks of 32 channels -> coalesced f32 stores
  #pragma unroll
  for (int c = 0; c < 4; ++c) {
    if ((wave & 3) == c) {
      #pragma unroll
      for (int mt = 0; mt < 2; ++mt)
        #pragma unroll
        for (int nt = 0; nt < 4; ++nt) {
          int ml = mt * 16 + ((lane >> 4) << 3);
          int n  = n_off + nt * 16 + (lane & 15);
          #pragma unroll
          for (int r = 0; r < 8; ++r) stage[(ml + r) * 128 + n] = acc[mt][nt][r];
        }
    }
    __syncthreads();
    {
      int row = t >> 3;                 // 0..31
      int col = (t & 7) * 16;           // 0..112
      int o = m0 + c * 32 + row;
      float bias = bp[o];
      float* op = out + (((size_t)b * 512 + o) * 128 + h) * 128 + col;
      #pragma unroll
      for (int j = 0; j < 4; ++j) {
        v4f v = *(const v4f*)(stage + row * 128 + col + j * 4);
        v = v + bias;
        *(v4f*)(op + j * 4) = v;
      }
    }
    __syncthreads();
  }
}

// ---------------------------------------------------------------- launch
extern "C" void kernel_launch(void* const* d_in, const int* in_sizes, int n_in,
                              void* d_out, int out_size, void* d_ws, size_t ws_size,
                              hipStream_t stream) {
  const float* x     = (const float*)d_in[0];  // (8,512,126,126)
  const float* wqkv  = (const float*)d_in[1];  // (1536,512)
  const float* wproj = (const float*)d_in[2];  // (512,512)
  const float* bproj = (const float*)d_in[3];  // (512)
  const float* btab  = (const float*)d_in[4];  // (225,16)

  char* ws = (char*)d_ws;
  __bf16* wq_bf = (__bf16*)ws;                                         // 1.5 MB
  __bf16* wp_bf = (__bf16*)(ws + (size_t)1536 * 512 * 2);              // 0.5 MB
  __bf16* qkv   = (__bf16*)(ws + (size_t)2097152);                     // 384 MB
  __bf16* abuf  = (__bf16*)(ws + (size_t)2097152 + 402653184);         // 128 MB

  cvt_weights<<<768, 256, 0, stream>>>(wqkv, wproj, wq_bf, wp_bf);
  qkv_gemm<<<dim3(128, 12, 8), 256, 0, stream>>>(x, wq_bf, qkv);
  attn_kernel<<<dim3(2048, 16), 128, 0, stream>>>(qkv, btab, abuf);
  proj_gemm<<<dim3(128, 4, 8), 256, 0, stream>>>(abuf, wp_bf, bproj, (float*)d_out);
}
